// HydraFusionHead_26259430048572
// MI455X (gfx1250) — matrix operand order repaired and checked by reading, not compiled, not run on hardware
//
#include <hip/hip_runtime.h>

// ---------------------------------------------------------------------------
// CDNA5 (gfx1250) wave32 WMMA implementation of the HydraFusionHead.
// All matmuls use v_wmma_f32_16x16x32_bf16. Activations staged NHWC bf16.
// Conv input halo staged in LDS via GLOBAL_LOAD_ASYNC_TO_LDS_B128 (ASYNCcnt).
// ---------------------------------------------------------------------------

typedef __bf16 bf16;
typedef bf16  v16bf __attribute__((ext_vector_type(16)));
typedef float v8f   __attribute__((ext_vector_type(8)));
typedef unsigned int u32;
typedef u32 v4u __attribute__((ext_vector_type(4)));

union Frag16 { v16bf v; v4u q[2]; };

#define B_   4
#define H_   128
#define W_   128
#define HW_  16384

static __device__ __forceinline__ v8f wmma_bf16(const Frag16& a, const Frag16& b, v8f c) {
  return __builtin_amdgcn_wmma_f32_16x16x32_bf16(false, a.v, false, b.v, (short)0, c, false, false);
}

// Sampler point offsets (pixel units), faithful to the reference's reshape quirks.
static __device__ const float OFFX[3][9] = {
  {-1.f, 0.f, 1.f, -1.f, 0.f, 1.f, -1.f, 0.f, 1.f},                                  // center
  {-1.f, 2.4375f, -1.f, -2.4375f, 1.f, 2.4375f, 1.f, -2.4375f, 0.f},                 // corner (8 used)
  {-1.625f, -1.625f, -1.625f, 0.f, -1.625f, 1.625f, 0.f, -1.625f, 0.f}               // fg
};
static __device__ const float OFFY[3][9] = {
  {-1.f, -1.f, -1.f, 0.f, 0.f, 0.f, 1.f, 1.f, 1.f},
  {2.4375f, -1.f, -2.4375f, -1.f, 2.4375f, 1.f, -2.4375f, 1.f, 0.f},
  {0.f, 0.f, 1.625f, 1.625f, -1.625f, 1.625f, 0.f, 1.625f, 1.625f}
};

// ------------------------------- pack / cast -------------------------------

__global__ __launch_bounds__(256) void cast_base_k(const float* __restrict__ src,
                                                   bf16* __restrict__ dst) {
  int i = blockIdx.x * 256 + threadIdx.x;           // over B*256*HW, NCHW order
  if (i >= B_ * 256 * HW_) return;
  int idx = i & (HW_ - 1);
  int c   = (i >> 14) & 255;
  int b   = i >> 22;
  dst[(size_t)((b << 14) + idx) * 256 + c] = (bf16)src[i];
}

__global__ __launch_bounds__(256) void pack_conv_w_k(const float* __restrict__ w,
                                                     bf16* __restrict__ dst) {
  int i = blockIdx.x * 256 + threadIdx.x;           // [64][2304], k = tap*256 + c
  if (i >= 64 * 2304) return;
  int o = i / 2304, k = i - o * 2304;
  int tap = k >> 8, c = k & 255;
  dst[i] = (bf16)w[(o * 256 + c) * 9 + tap];
}

__global__ __launch_bounds__(256) void pack_head_w_k(const float* __restrict__ w,
                                                     bf16* __restrict__ dst, int hc) {
  int i = blockIdx.x * 256 + threadIdx.x;           // [16][576], k = tap*64 + c
  if (i >= 16 * 576) return;
  int o = i / 576, k = i - o * 576;
  int tap = k >> 6, c = k & 63;
  dst[i] = (bf16)((o < hc) ? w[(o * 64 + c) * 9 + tap] : 0.f);
}

__global__ __launch_bounds__(256) void pack_lin_w_k(const float* __restrict__ w,
                                                    bf16* __restrict__ dst,
                                                    int M, int Cin, int Kpad) {
  int i = blockIdx.x * 256 + threadIdx.x;           // dst [M][Kpad], w [M][Cin]
  if (i >= M * Kpad) return;
  int r = i / Kpad, k = i - r * Kpad;
  dst[i] = (bf16)((k < Cin) ? w[(size_t)r * Cin + k] : 0.f);
}

// ------------------------------ conv 3x3 (WMMA) ----------------------------
// Implicit GEMM: M=64 (4 tiles = 4 waves), N=16 pixels/workgroup, K=9*256.
// The 3x18x256 input halo is staged into LDS with async global->LDS copies,
// then all 4 waves consume it from LDS (9x global traffic reduction).

__global__ __launch_bounds__(128) void conv3x3_k(const bf16* __restrict__ baseH,
                                                 const bf16* __restrict__ Wc,
                                                 const float* __restrict__ cb,
                                                 const float* __restrict__ cs,
                                                 const float* __restrict__ ct,
                                                 bf16* __restrict__ featH) {
  extern __shared__ bf16 tile[];              // [3][18][256] bf16 = 27648 B @ LDS off 0
  __builtin_amdgcn_s_wait_tensorcnt(0);
  const int tid  = threadIdx.x;
  const int lane = tid & 31;
  const int mt   = tid >> 5;                  // wave = M tile 0..3
  const int pix0 = blockIdx.x * 16;           // segment start over B*HW
  const int b    = pix0 >> 14;
  const int idx0 = pix0 & (HW_ - 1);
  const int y    = idx0 >> 7;
  const int x0   = idx0 & (W_ - 1);

  // ---- stage halo: chunks of 8 channels (16B); 3*18*32 = 1728 chunks ----
  for (int ci = tid; ci < 1728; ci += 128) {
    int cg = ci & 31;                         // channel group (8 ch)
    int t2 = ci >> 5;                         // 0..53
    int px = t2 % 18;                         // halo x: 0..17
    int r  = t2 / 18;                         // halo y: 0..2
    int yy = y + r - 1;
    int xx = x0 + px - 1;
    unsigned ldsoff = (unsigned)(((r * 18 + px) * 256 + cg * 8) * 2);
    if (((unsigned)yy < (unsigned)H_) && ((unsigned)xx < (unsigned)W_)) {
      const bf16* g = baseH + ((size_t)((b << 14) + (yy << 7) + xx) << 8) + cg * 8;
      asm volatile("global_load_async_to_lds_b128 %0, %1, off"
                   :: "v"(ldsoff), "v"(g) : "memory");
    } else {
      *(v4u*)(tile + ((r * 18 + px) * 256 + cg * 8)) = (v4u)0;
    }
  }
  asm volatile("s_wait_asynccnt 0x0" ::: "memory");
  __syncthreads();

  // ---- WMMA main loop ----
  const int hi = lane >> 4;
  const int lx = lane & 15;
  v8f acc = {};
  const bf16* wrow = Wc + (size_t)(mt * 16 + lx) * 2304 + hi * 16;
  for (int tap = 0; tap < 9; ++tap) {
    const int r   = tap / 3;                  // dy+1
    const int dxp = tap - r * 3;              // dx+1
    const bf16* lsrc = tile + ((r * 18 + lx + dxp) * 256) + hi * 16;
    const bf16* wt   = wrow + tap * 256;
    __builtin_prefetch(wt + 256, 0, 1);
    for (int cbk = 0; cbk < 256; cbk += 32) {
      Frag16 a, bb;
      a.q[0]  = *(const v4u*)(wt + cbk);
      a.q[1]  = *(const v4u*)(wt + cbk + 8);
      bb.q[0] = *(const v4u*)(lsrc + cbk);
      bb.q[1] = *(const v4u*)(lsrc + cbk + 8);
      acc = wmma_bf16(a, bb, acc);
    }
  }
  bf16* orow = featH + ((size_t)(pix0 + lx) << 6);
  const int chb = mt * 16 + hi * 8;
#pragma unroll
  for (int r = 0; r < 8; ++r) {
    int ch = chb + r;
    float v = (acc[r] + cb[ch]) * cs[ch] + ct[ch];
    orow[ch] = (bf16)(v > 0.f ? v : 0.f);
  }
}

// ---------------------------- head conv 3x3 (WMMA) -------------------------
// M=16 (hc valid), K=9*64. Writes fp32 map to d_out and bf16 NHWC map buffer.

__global__ __launch_bounds__(256) void headconv_k(const bf16* __restrict__ featH,
                                                  const bf16* __restrict__ Wh,
                                                  const float* __restrict__ hb,
                                                  float* __restrict__ dout,
                                                  bf16* __restrict__ mapH,
                                                  int hc, int outChBase) {
  const int lane = threadIdx.x & 31;
  const int wid  = (blockIdx.x * 256 + threadIdx.x) >> 5;
  const int hi   = lane >> 4;
  const int pix  = wid * 16 + (lane & 15);
  const int b    = pix >> 14;
  const int idx  = pix & (HW_ - 1);
  const int x = idx & (W_ - 1), y = idx >> 7;
  v8f acc = {};
  const bf16* wrow = Wh + (size_t)(lane & 15) * 576 + hi * 16;
  for (int tap = 0; tap < 9; ++tap) {
    const int dy = tap / 3 - 1, dx = tap - (tap / 3) * 3 - 1;
    const int yy = y + dy, xx = x + dx;
    const bool inb = ((unsigned)yy < (unsigned)H_) && ((unsigned)xx < (unsigned)W_);
    const bf16* src = featH + ((size_t)((b << 14) + (yy << 7) + xx) << 6) + hi * 16;
    const bf16* wt  = wrow + tap * 64;
    for (int cbk = 0; cbk < 64; cbk += 32) {
      Frag16 a, bb;
      a.q[0] = *(const v4u*)(wt + cbk);
      a.q[1] = *(const v4u*)(wt + cbk + 8);
      if (inb) {
        bb.q[0] = *(const v4u*)(src + cbk);
        bb.q[1] = *(const v4u*)(src + cbk + 8);
      } else {
        bb.q[0] = (v4u)0; bb.q[1] = (v4u)0;
      }
      acc = wmma_bf16(a, bb, acc);
    }
  }
  bf16* mrow = mapH + ((size_t)pix << 4);
#pragma unroll
  for (int r = 0; r < 8; ++r) {
    int o = hi * 8 + r;
    if (o < hc) {
      float v = acc[r] + hb[o];
      mrow[o] = (bf16)v;
      dout[(size_t)(b * 26 + outChBase + o) * HW_ + idx] = v;
    } else {
      mrow[o] = (bf16)0.f;
    }
  }
}

// ------------------------------- sampling ----------------------------------
// Builds S[pix][640] bf16 (k = c*npts + p; geo channels; zero pad to 640).

__global__ __launch_bounds__(256) void sample_k(const bf16* __restrict__ featHb,
                                                const bf16* __restrict__ mapHb,
                                                bf16* __restrict__ S,
                                                int npts, int hc, int srcsel) {
  int t = blockIdx.x * 256 + threadIdx.x;
  if (t >= HW_ * npts) return;
  int idx = t / npts;
  int p   = t - idx * npts;
  int x = idx & (W_ - 1), y = idx >> 7;
  float gx = x + 0.5f + OFFX[srcsel][p];
  float gy = y + 0.5f + OFFY[srcsel][p];
  float sx = gx - 0.5f, sy = gy - 0.5f;
  float fx0 = floorf(sx), fy0 = floorf(sy);
  int ix0 = (int)fx0, iy0 = (int)fy0;
  int ix1 = ix0 + 1, iy1 = iy0 + 1;
  float dx = sx - fx0, dy = sy - fy0;
  float vx0 = ((unsigned)ix0 < (unsigned)W_) ? 1.f : 0.f;
  float vx1 = ((unsigned)ix1 < (unsigned)W_) ? 1.f : 0.f;
  float vy0 = ((unsigned)iy0 < (unsigned)H_) ? 1.f : 0.f;
  float vy1 = ((unsigned)iy1 < (unsigned)H_) ? 1.f : 0.f;
  float w00 = (1.f - dx) * (1.f - dy) * vx0 * vy0;
  float w10 = dx * (1.f - dy) * vx1 * vy0;
  float w01 = (1.f - dx) * dy * vx0 * vy1;
  float w11 = dx * dy * vx1 * vy1;
  int cx0 = min(max(ix0, 0), W_ - 1), cx1 = min(max(ix1, 0), W_ - 1);
  int cy0 = min(max(iy0, 0), H_ - 1), cy1 = min(max(iy1, 0), H_ - 1);
  const bf16* f00 = featHb + ((size_t)((cy0 << 7) + cx0) << 6);
  const bf16* f10 = featHb + ((size_t)((cy0 << 7) + cx1) << 6);
  const bf16* f01 = featHb + ((size_t)((cy1 << 7) + cx0) << 6);
  const bf16* f11 = featHb + ((size_t)((cy1 << 7) + cx1) << 6);
  const bf16* m00 = mapHb + ((size_t)((cy0 << 7) + cx0) << 4);
  const bf16* m10 = mapHb + ((size_t)((cy0 << 7) + cx1) << 4);
  const bf16* m01 = mapHb + ((size_t)((cy1 << 7) + cx0) << 4);
  const bf16* m11 = mapHb + ((size_t)((cy1 << 7) + cx1) << 4);
  bf16* srow = S + (size_t)idx * 640;
  for (int c = 0; c < 64; ++c) {
    float v = w00 * (float)f00[c] + w10 * (float)f10[c]
            + w01 * (float)f01[c] + w11 * (float)f11[c];
    srow[c * npts + p] = (bf16)v;
  }
  for (int c = 0; c < hc; ++c) {
    float v = w00 * (float)m00[c] + w10 * (float)m10[c]
            + w01 * (float)m01[c] + w11 * (float)m11[c];
    srow[(64 + c) * npts + p] = (bf16)v;
  }
  int C = 64 + hc;
  srow[C * npts + p]       = (bf16)(gx * (1.f / 128.f));
  srow[(C + 1) * npts + p] = (bf16)(gy * (1.f / 128.f));
  if (p == 0) {
    for (int k = (C + 2) * npts; k < 640; ++k) srow[k] = (bf16)0.f;
  }
}

// ---------------------- generic 16-out-ch GEMM + CBR (WMMA) ----------------
// out[n][pix][16] = relu((W[n] @ X[pix] + b) * s + t); waves: (head, pixtile).

__global__ __launch_bounds__(256) void gemm16_k(const bf16* __restrict__ X, int rowLen,
                                                const bf16* __restrict__ Wt,
                                                const float* __restrict__ gb,
                                                const float* __restrict__ gs,
                                                const float* __restrict__ gt,
                                                float* __restrict__ out) {
  const int lane = threadIdx.x & 31;
  const int wid  = (blockIdx.x * 256 + threadIdx.x) >> 5;
  const int n    = wid & 3;
  const int tile = wid >> 2;                 // 0..1023 over HW
  const int hi   = lane >> 4;
  const int pix  = tile * 16 + (lane & 15);
  v8f acc = {};
  const bf16* wrow = Wt + (size_t)(n * 16 + (lane & 15)) * rowLen + hi * 16;
  const bf16* xrow = X + (size_t)pix * rowLen + hi * 16;
  for (int kb = 0; kb < rowLen; kb += 32) {
    Frag16 a, bb;
    a.q[0]  = *(const v4u*)(wrow + kb);
    a.q[1]  = *(const v4u*)(wrow + kb + 8);
    bb.q[0] = *(const v4u*)(xrow + kb);
    bb.q[1] = *(const v4u*)(xrow + kb + 8);
    acc = wmma_bf16(a, bb, acc);
  }
  float* orow = out + ((size_t)(n * HW_ + pix) << 4) + hi * 8;
#pragma unroll
  for (int r = 0; r < 8; ++r) {
    int cg = n * 16 + hi * 8 + r;
    float v = (acc[r] + gb[cg]) * gs[cg] + gt[cg];
    orow[r] = v > 0.f ? v : 0.f;
  }
}

// ------------------------------- attention ---------------------------------

__global__ __launch_bounds__(256) void attn_k(const float* __restrict__ Q,
                                              const float* __restrict__ KV,
                                              bf16* __restrict__ att) {
  int t = blockIdx.x * 256 + threadIdx.x;    // over 4*HW
  if (t >= 4 * HW_) return;
  int n = t >> 14;
  int idx = t & (HW_ - 1);
  const float* q  = Q  + ((size_t)(n * HW_ + idx) << 4);
  const float* kc = KV + ((size_t)((0 * 4 + n) * HW_ + idx) << 4);
  const float* vc = KV + ((size_t)((1 * 4 + n) * HW_ + idx) << 4);
  const float* ko = KV + ((size_t)((2 * 4 + n) * HW_ + idx) << 4);
  const float* vo = KV + ((size_t)((3 * 4 + n) * HW_ + idx) << 4);
  const float* kf = KV + ((size_t)((4 * 4 + n) * HW_ + idx) << 4);
  const float* vf = KV + ((size_t)((5 * 4 + n) * HW_ + idx) << 4);
  float l0 = 0.f, l1 = 0.f, l2 = 0.f;
#pragma unroll
  for (int j = 0; j < 16; ++j) {
    float qj = q[j];
    l0 += qj * kc[j]; l1 += qj * ko[j]; l2 += qj * kf[j];
  }
  float m = fmaxf(l0, fmaxf(l1, l2));
  float e0 = __expf(l0 - m), e1 = __expf(l1 - m), e2 = __expf(l2 - m);
  float inv = 1.f / (e0 + e1 + e2);
  bf16* a = att + ((size_t)(n * HW_ + idx) << 4);
#pragma unroll
  for (int j = 0; j < 16; ++j)
    a[j] = (bf16)((e0 * vc[j] + e1 * vo[j] + e2 * vf[j]) * inv);
}

// --------------------------- predict p1 (WMMA) + p2 ------------------------
// K = [base(256) | att(16) | pad(16)] = 288.  p2 reduced through LDS.

__global__ __launch_bounds__(256) void p1p2_k(const bf16* __restrict__ baseHb,
                                              const bf16* __restrict__ att,
                                              const bf16* __restrict__ Wp1,
                                              const float* __restrict__ b1,
                                              const float* __restrict__ s1,
                                              const float* __restrict__ t1,
                                              const float* __restrict__ w2,
                                              const float* __restrict__ b2,
                                              float* __restrict__ dout, int b) {
  __shared__ float lds[8][16][68];
  const int lane = threadIdx.x & 31;
  const int wv   = threadIdx.x >> 5;
  const int wid  = blockIdx.x * 8 + wv;
  const int n    = wid & 3;
  const int tile = wid >> 2;                 // 0..1023 over HW
  const int hi   = lane >> 4;
  const int pix  = tile * 16 + (lane & 15);
  const bf16* xrow = baseHb + ((size_t)pix << 8) + hi * 16;
  const bf16* arow = att + ((size_t)(n * HW_ + pix) << 4);
  for (int mt = 0; mt < 4; ++mt) {
    v8f acc = {};
    const bf16* wrow = Wp1 + (size_t)(n * 64 + mt * 16 + (lane & 15)) * 288 + hi * 16;
    for (int kc = 0; kc < 9; ++kc) {
      Frag16 a, bb;
      a.q[0] = *(const v4u*)(wrow + kc * 32);
      a.q[1] = *(const v4u*)(wrow + kc * 32 + 8);
      if (kc < 8) {
        bb.q[0] = *(const v4u*)(xrow + kc * 32);
        bb.q[1] = *(const v4u*)(xrow + kc * 32 + 8);
      } else if (hi == 0) {
        bb.q[0] = *(const v4u*)(arow);
        bb.q[1] = *(const v4u*)(arow + 8);
      } else {
        bb.q[0] = (v4u)0; bb.q[1] = (v4u)0;
      }
      acc = wmma_bf16(a, bb, acc);
    }
#pragma unroll
    for (int r = 0; r < 8; ++r) {
      int ch = mt * 16 + hi * 8 + r;
      int cg = n * 64 + ch;
      float v = (acc[r] + b1[cg]) * s1[cg] + t1[cg];
      lds[wv][lane & 15][ch] = v > 0.f ? v : 0.f;
    }
  }
  __syncthreads();
  int p  = lane >> 4;
  int px = lane & 15;
  float s = b2[n * 2 + p];
  const float* wr = w2 + (n * 2 + p) * 64;
  for (int o = 0; o < 64; ++o) s += wr[o] * lds[wv][px][o];
  dout[(size_t)(b * 26 + 18 + n * 2 + p) * HW_ + tile * 16 + px] = s;
}

// ------------------------------- launch ------------------------------------

extern "C" void kernel_launch(void* const* d_in, const int* in_sizes, int n_in,
                              void* d_out, int out_size, void* d_ws, size_t ws_size,
                              hipStream_t stream) {
  (void)in_sizes; (void)n_in; (void)out_size; (void)ws_size;
  char* ws = (char*)d_ws;

  // workspace layout (bytes)
  constexpr size_t SZ_BASEH = (size_t)B_ * HW_ * 256 * 2;   // NHWC bf16
  constexpr size_t SZ_FEATH = (size_t)3 * B_ * HW_ * 64 * 2;
  constexpr size_t SZ_MAPH  = (size_t)3 * B_ * HW_ * 16 * 2;
  constexpr size_t SZ_WC    = (size_t)3 * 64 * 2304 * 2;
  constexpr size_t SZ_WH    = (size_t)3 * 16 * 576 * 2;
  constexpr size_t SZ_WQ    = (size_t)64 * 256 * 2;
  constexpr size_t SZ_WKV   = (size_t)6 * 64 * 640 * 2;
  constexpr size_t SZ_WP1   = (size_t)256 * 288 * 2;
  constexpr size_t SZ_S     = (size_t)HW_ * 640 * 2;        // per-b reuse
  constexpr size_t SZ_KV    = (size_t)6 * 4 * HW_ * 16 * 4; // per-b
  constexpr size_t SZ_Q     = (size_t)4 * HW_ * 16 * 4;     // per-b

  size_t off = 0;
  bf16* baseH = (bf16*)(ws + off); off += SZ_BASEH;
  bf16* featH = (bf16*)(ws + off); off += SZ_FEATH;
  bf16* mapH  = (bf16*)(ws + off); off += SZ_MAPH;
  bf16* Wc    = (bf16*)(ws + off); off += SZ_WC;
  bf16* Wh    = (bf16*)(ws + off); off += SZ_WH;
  bf16* Wq    = (bf16*)(ws + off); off += SZ_WQ;
  bf16* Wkv   = (bf16*)(ws + off); off += SZ_WKV;
  bf16* Wp1   = (bf16*)(ws + off); off += SZ_WP1;
  bf16* Sbuf  = (bf16*)(ws + off); off += SZ_S;
  float* KVb  = (float*)(ws + off); off += SZ_KV;
  float* Qb   = (float*)(ws + off); off += SZ_Q;
  bf16* Attb  = (bf16*)(ws + off);

  const float* base = (const float*)d_in[0];
  float* dout = (float*)d_out;

  const int hcs[3]   = {3, 12, 3};
  const int npts[3]  = {9, 8, 9};
  const int obase[3] = {0, 3, 15};
  const int Cin[6]   = {621, 621, 624, 624, 621, 621};

  // 1) cast base to NHWC bf16
  cast_base_k<<<(B_ * 256 * HW_) / 256, 256, 0, stream>>>(base, baseH);

  // 2) pack all weights
  for (int s3 = 0; s3 < 3; ++s3) {
    pack_conv_w_k<<<(64 * 2304 + 255) / 256, 256, 0, stream>>>(
        (const float*)d_in[1 + 6 * s3], Wc + (size_t)s3 * 64 * 2304);
    pack_head_w_k<<<(16 * 576 + 255) / 256, 256, 0, stream>>>(
        (const float*)d_in[5 + 6 * s3], Wh + (size_t)s3 * 16 * 576, hcs[s3]);
  }
  pack_lin_w_k<<<(64 * 256 + 255) / 256, 256, 0, stream>>>(
      (const float*)d_in[19], Wq, 64, 256, 256);
  for (int t = 0; t < 6; ++t)
    pack_lin_w_k<<<(64 * 640 + 255) / 256, 256, 0, stream>>>(
        (const float*)d_in[23 + 4 * t], Wkv + (size_t)t * 64 * 640, 64, Cin[t], 640);
  pack_lin_w_k<<<(256 * 288 + 255) / 256, 256, 0, stream>>>(
      (const float*)d_in[47], Wp1, 256, 272, 288);

  // 3) three CBR convs (LDS-staged, async copies) + head convs
  for (int s3 = 0; s3 < 3; ++s3) {
    conv3x3_k<<<B_ * HW_ / 16, 128, 3 * 18 * 256 * 2, stream>>>(
        baseH, Wc + (size_t)s3 * 64 * 2304,
        (const float*)d_in[2 + 6 * s3], (const float*)d_in[3 + 6 * s3],
        (const float*)d_in[4 + 6 * s3],
        featH + (size_t)s3 * B_ * HW_ * 64);
    headconv_k<<<512, 256, 0, stream>>>(
        featH + (size_t)s3 * B_ * HW_ * 64, Wh + (size_t)s3 * 16 * 576,
        (const float*)d_in[6 + 6 * s3], dout,
        mapH + (size_t)s3 * B_ * HW_ * 16, hcs[s3], obase[s3]);
  }

  // 4) per-batch: sample -> K/V gemms -> q -> attention -> predict
  for (int b = 0; b < B_; ++b) {
    const bf16* baseHb = baseH + (size_t)b * HW_ * 256;
    for (int s3 = 0; s3 < 3; ++s3) {
      int nthr = HW_ * npts[s3];
      sample_k<<<(nthr + 255) / 256, 256, 0, stream>>>(
          featH + (size_t)(s3 * B_ + b) * HW_ * 64,
          mapH + (size_t)(s3 * B_ + b) * HW_ * 16,
          Sbuf, npts[s3], hcs[s3], s3);
      for (int kv = 0; kv < 2; ++kv) {
        int t = s3 * 2 + kv;
        gemm16_k<<<512, 256, 0, stream>>>(
            Sbuf, 640, Wkv + (size_t)t * 64 * 640,
            (const float*)d_in[24 + 4 * t], (const float*)d_in[25 + 4 * t],
            (const float*)d_in[26 + 4 * t],
            KVb + (size_t)t * 4 * HW_ * 16);
      }
    }
    gemm16_k<<<512, 256, 0, stream>>>(
        baseHb, 256, Wq,
        (const float*)d_in[20], (const float*)d_in[21], (const float*)d_in[22], Qb);
    attn_k<<<(4 * HW_) / 256, 256, 0, stream>>>(Qb, KVb, Attb);
    p1p2_k<<<512, 256, 0, stream>>>(
        baseHb, Attb, Wp1,
        (const float*)d_in[48], (const float*)d_in[49], (const float*)d_in[50],
        (const float*)d_in[51], (const float*)d_in[52], dout, b);
  }
}
